// CRF_2284922601504
// MI455X (gfx1250) — matrix operand order repaired and checked by reading, compiled, not run on hardware
//
#include <hip/hip_runtime.h>
#include <hip/hip_bf16.h>

typedef _Float16 v8h  __attribute__((ext_vector_type(8)));
typedef _Float16 v16h __attribute__((ext_vector_type(16)));
typedef float    v8f  __attribute__((ext_vector_type(8)));
typedef unsigned int u32x4 __attribute__((ext_vector_type(4)));
typedef unsigned int u32x8 __attribute__((ext_vector_type(8)));

#define NTAG 64
#define LOG2E 1.4426950408889634f
#define LN2   0.6931471805599453f

// One full scan step: alpha' = (mx + feat) + ln2*log2(E^T @ exp2((alpha-mx)*log2e))
// d[] is WMMA D layout: lane = batch col (n), row j = 16*jb + 8*g + v.
__device__ __forceinline__ void crf_step(v8f d[4], const v16h ET[4][2],
                                         const float4 fr[4][2], const int g) {
  // 1. per-batch running max (batch lives in lanes L and L^16)
  float m1 = d[0][0];
#pragma unroll
  for (int jb = 0; jb < 4; ++jb)
#pragma unroll
    for (int v = 0; v < 8; ++v) m1 = fmaxf(m1, d[jb][v]);
  const float mx = fmaxf(m1, __shfl_xor(m1, 16, 32));
  const float nmxl = -mx * LOG2E;   // fold bias into the exp2 fma

  // 2. P = exp2(alpha*log2e - mx*log2e) packed to f16x2 dwords per tile
  u32x4 P[4];
#pragma unroll
  for (int tt = 0; tt < 4; ++tt) {
#pragma unroll
    for (int v = 0; v < 8; v += 2) {
      const float e0 =
          __builtin_amdgcn_exp2f(__builtin_fmaf(d[tt][v],     LOG2E, nmxl));
      const float e1 =
          __builtin_amdgcn_exp2f(__builtin_fmaf(d[tt][v + 1], LOG2E, nmxl));
      P[tt][v >> 1] =
          __builtin_bit_cast(unsigned int, __builtin_amdgcn_cvt_pkrtz(e0, e1));
    }
  }

  // 3. D-layout -> B-operand: swap row halves between lanes L and L^16
  v16h bf[2];
#pragma unroll
  for (int kb = 0; kb < 2; ++kb) {
    u32x4 send = g ? P[2 * kb] : P[2 * kb + 1];
    u32x4 recv;
#pragma unroll
    for (int dd = 0; dd < 4; ++dd)
      recv[dd] = (unsigned)__shfl_xor((int)send[dd], 16, 32);
    u32x8 f;
#pragma unroll
    for (int dd = 0; dd < 4; ++dd) {
      f[dd]     = g ? recv[dd]          : P[2 * kb][dd];
      f[dd + 4] = g ? P[2 * kb + 1][dd] : recv[dd];
    }
    bf[kb] = __builtin_bit_cast(v16h, f);
  }

  // 4. 8x v_wmma_f32_16x16x32_f16 + log2 reconstruction
#pragma unroll
  for (int jb = 0; jb < 4; ++jb) {
    v8f acc = {0.f, 0.f, 0.f, 0.f, 0.f, 0.f, 0.f, 0.f};
    acc = __builtin_amdgcn_wmma_f32_16x16x32_f16(
        false, ET[jb][0], false, bf[0], (short)0, acc, false, false);
    acc = __builtin_amdgcn_wmma_f32_16x16x32_f16(
        false, ET[jb][1], false, bf[1], (short)0, acc, false, false);
    float fv[8];
    fv[0] = fr[jb][0].x; fv[1] = fr[jb][0].y; fv[2] = fr[jb][0].z; fv[3] = fr[jb][0].w;
    fv[4] = fr[jb][1].x; fv[5] = fr[jb][1].y; fv[6] = fr[jb][1].z; fv[7] = fr[jb][1].w;
#pragma unroll
    for (int v = 0; v < 8; ++v)
      d[jb][v] = __builtin_fmaf(LN2, __builtin_amdgcn_logf(acc[v]), mx + fv[v]);
  }
}

__device__ __forceinline__ void load_feats(float4 dst[4][2], const float* base,
                                           const int t, const int g) {
#pragma unroll
  for (int jb = 0; jb < 4; ++jb) {
    const float4* p = (const float4*)(base + (size_t)t * NTAG + 16 * jb + 8 * g);
    dst[jb][0] = p[0];
    dst[jb][1] = p[1];
  }
}

// One wave (32 lanes) advances the CRF forward recurrence for 16 batches.
__global__ __launch_bounds__(32) void crf_forward_kernel(
    const float* __restrict__ feats,      // [B,S,64]
    const float* __restrict__ trans,      // [64,64]
    float* __restrict__ fwd,              // [B] forward scores
    int S) {
  const int lane = threadIdx.x;
  const int n    = lane & 15;   // batch column within group / D column
  const int g    = lane >> 4;   // lane half-group
  const int bg   = blockIdx.x;  // 16-batch group

  // ---- E^T A-fragments, f16, register resident for the whole scan ----
  // A tile (jb,kb): A[m][k] = exp(trans[32*kb+k][16*jb+m]); m = n,
// A-layout slot s -> k = s + 16*g.
  v16h ET[4][2];
#pragma unroll
  for (int jb = 0; jb < 4; ++jb)
#pragma unroll
    for (int kb = 0; kb < 2; ++kb)
#pragma unroll
      for (int s = 0; s < 16; ++s) {
        const int k = s + 16 * g;
        ET[jb][kb][s] = (_Float16)__builtin_amdgcn_exp2f(
            LOG2E * trans[(32 * kb + k) * NTAG + 16 * jb + n]);
      }

  const float* fbase = feats + (size_t)(bg * 16 + n) * S * NTAG;

  // ---- alpha(t=0) = feats[:,0,:] straight into D layout ----
  v8f d[4];
#pragma unroll
  for (int jb = 0; jb < 4; ++jb) {
    const float4* p = (const float4*)(fbase + 16 * jb + 8 * g);
    const float4 a = p[0], b = p[1];
    d[jb][0] = a.x; d[jb][1] = a.y; d[jb][2] = a.z; d[jb][3] = a.w;
    d[jb][4] = b.x; d[jb][5] = b.y; d[jb][6] = b.z; d[jb][7] = b.w;
  }

  // ---- scan, unrolled by 2 with ping-pong feats buffers, prefetch dist 2 ----
  float4 fX[4][2], fY[4][2];
  load_feats(fX, fbase, 1, g);
  if (S > 2) load_feats(fY, fbase, 2, g);

  int t = 1;
  for (; t + 1 < S; t += 2) {
    crf_step(d, ET, fX, g);                     // consumes feats[t]
    if (t + 2 < S) load_feats(fX, fbase, t + 2, g);
    crf_step(d, ET, fY, g);                     // consumes feats[t+1]
    if (t + 3 < S) load_feats(fY, fbase, t + 3, g);
  }
  if (t < S) crf_step(d, ET, fX, g);            // odd leftover step

  // ---- forward_score = logsumexp_j alpha ----
  float m2 = d[0][0];
#pragma unroll
  for (int jb = 0; jb < 4; ++jb)
#pragma unroll
    for (int v = 0; v < 8; ++v) m2 = fmaxf(m2, d[jb][v]);
  m2 = fmaxf(m2, __shfl_xor(m2, 16, 32));
  const float nm2l = -m2 * LOG2E;
  float s2 = 0.f;
#pragma unroll
  for (int jb = 0; jb < 4; ++jb)
#pragma unroll
    for (int v = 0; v < 8; ++v)
      s2 += __builtin_amdgcn_exp2f(__builtin_fmaf(d[jb][v], LOG2E, nm2l));
  s2 += __shfl_xor(s2, 16, 32);
  if (g == 0)
    fwd[bg * 16 + n] = __builtin_fmaf(LN2, __builtin_amdgcn_logf(s2), m2);
}

// gold_score[b] = sum_t feats[b,t,tag] + sum_t trans[tag_{t-1},tag_t]
__global__ __launch_bounds__(256) void crf_gold_kernel(
    const float* __restrict__ feats, const int* __restrict__ tags,
    const float* __restrict__ trans, float* __restrict__ gold, int B, int S) {
  const int b = blockIdx.x * blockDim.x + threadIdx.x;
  if (b >= B) return;
  const int* tg = tags + (size_t)b * S;
  const float* fb = feats + (size_t)b * S * NTAG;
  int prev = tg[0];
  float e = fb[prev];
  float tr = 0.f;
  for (int t = 1; t < S; ++t) {
    const int cur = tg[t];
    e += fb[(size_t)t * NTAG + cur];
    tr += trans[prev * NTAG + cur];
    prev = cur;
  }
  gold[b] = e + tr;
}

// deterministic scalar reduction: mean(fwd - gold)
__global__ __launch_bounds__(256) void crf_final_kernel(
    const float* __restrict__ fwd, const float* __restrict__ gold,
    float* __restrict__ out, int B) {
  __shared__ float sm[256];
  float s = 0.f;
  for (int i = threadIdx.x; i < B; i += 256) s += fwd[i] - gold[i];
  sm[threadIdx.x] = s;
  __syncthreads();
  for (int k = 128; k > 0; k >>= 1) {
    if ((int)threadIdx.x < k) sm[threadIdx.x] += sm[threadIdx.x + k];
    __syncthreads();
  }
  if (threadIdx.x == 0) out[0] = sm[0] / (float)B;
}

extern "C" void kernel_launch(void* const* d_in, const int* in_sizes, int n_in,
                              void* d_out, int out_size, void* d_ws, size_t ws_size,
                              hipStream_t stream) {
  const float* feats = (const float*)d_in[0];
  const int*   tags  = (const int*)d_in[1];
  const float* trans = (const float*)d_in[2];
  float* out = (float*)d_out;

  const int S = 512;
  const int B = in_sizes[1] / S;   // feats: [B,S,64], tags: [B,S]

  float* fwd  = (float*)d_ws;      // [B]
  float* gold = fwd + B;           // [B]

  crf_forward_kernel<<<B / 16, 32, 0, stream>>>(feats, trans, fwd, S);
  crf_gold_kernel<<<(B + 255) / 256, 256, 0, stream>>>(feats, tags, trans, gold, B, S);
  crf_final_kernel<<<1, 256, 0, stream>>>(fwd, gold, out, B);
}
